// ResNetBlock_TNN_86208583565596
// MI455X (gfx1250) — 8-bit matrix operand order repaired and checked by reading, compiled, not run on hardware
//
#include <hip/hip_runtime.h>
#include <cstdint>

#define LRELU_SLOPE 0.01f
#define BN_EPS 1e-5f

typedef int v8i __attribute__((ext_vector_type(8)));

__device__ __forceinline__ float ternf(float t, float d) {
  float s = (t > 0.f) ? 1.f : ((t < 0.f) ? -1.f : 0.f);
  return (fabsf(t) < d) ? 0.f : s;
}

__device__ __forceinline__ v8i zero_v8i() {
  v8i z;
#pragma unroll
  for (int i = 0; i < 8; ++i) z[i] = 0;
  return z;
}

__device__ __forceinline__ v8i wmma_iu8(v8i a, v8i b, v8i c) {
  return __builtin_amdgcn_wmma_i32_16x16x64_iu8(true, a, true, b, c, false, false);
}

__device__ __forceinline__ v8i make_a(uint64_t q0, uint64_t q1, uint64_t q2, uint64_t q3) {
  v8i a;
  a[0] = (int)(uint32_t)q0; a[1] = (int)(uint32_t)(q0 >> 32);
  a[2] = (int)(uint32_t)q1; a[3] = (int)(uint32_t)(q1 >> 32);
  a[4] = (int)(uint32_t)q2; a[5] = (int)(uint32_t)(q2 >> 32);
  a[6] = (int)(uint32_t)q3; a[7] = (int)(uint32_t)(q3 >> 32);
  return a;
}

__global__ void maxabs_kernel(const float* __restrict__ w, int n, float* __restrict__ dout) {
  __shared__ float red[256];
  float m = 0.f;
  for (int i = threadIdx.x; i < n; i += 256) m = fmaxf(m, fabsf(w[i]));
  red[threadIdx.x] = m;
  __syncthreads();
  for (int s = 128; s > 0; s >>= 1) {
    if ((int)threadIdx.x < s) red[threadIdx.x] = fmaxf(red[threadIdx.x], red[threadIdx.x + s]);
    __syncthreads();
  }
  if (threadIdx.x == 0) *dout = 0.1f * red[0];
}

__global__ void pack_w_kernel(const float* __restrict__ w, const float* __restrict__ dptr,
                              int8_t* __restrict__ out, int C, int nchunks) {
  float d = *dptr;
  int total = nchunks * 4096;
  for (int idx = blockIdx.x * blockDim.x + threadIdx.x; idx < total;
       idx += gridDim.x * blockDim.x) {
    int byte = idx & 31;
    int lane = (idx >> 5) & 31;
    int nt   = (idx >> 10) & 3;
    int kc   = idx >> 12;
    int v = byte >> 2, bb = byte & 3;
    int half = lane >> 4;
    int kk = (byte >> 3) * 16 + half * 8 + (byte & 7);
    int k = kc * 64 + kk;
    int n = nt * 16 + (lane & 15);
    int tap = k / C, c = k % C;
    int8_t val = 0;
    if (tap < 9) val = (int8_t)ternf(w[(n * C + c) * 9 + tap], d);
    out[idx] = val;
  }
}

__global__ __launch_bounds__(256) void prep_act_kernel(
    const float* __restrict__ x,
    const float* __restrict__ d11p, const float* __restrict__ d1p,
    const float* __restrict__ gamma_p, const float* __restrict__ beta_p,
    const float* __restrict__ mean_p, const float* __restrict__ var_p,
    int8_t* __restrict__ xt, int8_t* __restrict__ a1) {
  __shared__ uint32_t lds_xt[64 * 8];
  __shared__ uint32_t lds_a1[64 * 8];
  const int blk = blockIdx.x;
  const int xc = blk & 3;
  const int y  = (blk >> 2) & 255;
  const int b  = blk >> 10;
  const int x0 = xc * 64;
  const int xx = threadIdx.x & 63;
  const int cg = threadIdx.x >> 6;
  const float d11 = *d11p, d1 = *d1p;

  uint32_t tlo = 0, thi = 0, alo = 0, ahi = 0;
#pragma unroll
  for (int q = 0; q < 8; ++q) {
    int c = cg * 8 + q;
    float f = x[(((size_t)(b * 32 + c) * 256 + y) * 256) + x0 + xx];
    float t = ternf(f, d11);
    float inv = gamma_p[c] * rsqrtf(var_p[c] + BN_EPS);
    float v = t * inv + (beta_p[c] - mean_p[c] * inv);
    v = (v > 0.f) ? v : LRELU_SLOPE * v;
    uint32_t tb = (uint32_t)((int)t) & 255u;
    uint32_t ab = (uint32_t)((int)ternf(v, d1)) & 255u;
    if (q < 4) { tlo |= tb << (8 * q);        alo |= ab << (8 * q); }
    else       { thi |= tb << (8 * (q - 4));  ahi |= ab << (8 * (q - 4)); }
  }
  lds_xt[xx * 8 + cg * 2]     = tlo;
  lds_xt[xx * 8 + cg * 2 + 1] = thi;
  lds_a1[xx * 8 + cg * 2]     = alo;
  lds_a1[xx * 8 + cg * 2 + 1] = ahi;
  __syncthreads();

  size_t base = (((size_t)(b * 256 + y) * 256) + x0) * 32;
  uint32_t* xtd = (uint32_t*)(xt + base);
  uint32_t* a1d = (uint32_t*)(a1 + base);
  xtd[threadIdx.x]       = lds_xt[threadIdx.x];
  xtd[threadIdx.x + 256] = lds_xt[threadIdx.x + 256];
  a1d[threadIdx.x]       = lds_a1[threadIdx.x];
  a1d[threadIdx.x + 256] = lds_a1[threadIdx.x + 256];
}

#define PW1 260
__global__ __launch_bounds__(512) void conv1_kernel(
    const int8_t* __restrict__ a1, const int8_t* __restrict__ wpack,
    const float* __restrict__ d1p, const float* __restrict__ d2p,
    const float* __restrict__ b1,
    const float* __restrict__ gamma1, const float* __restrict__ beta1,
    const float* __restrict__ mean1, const float* __restrict__ var1,
    int8_t* __restrict__ a2) {
  __shared__ int8_t patch[3 * PW1 * 32];
  const int y = blockIdx.x & 255;
  const int b = blockIdx.x >> 8;
  const int tid = threadIdx.x;
  for (int i = tid; i < 3 * PW1 * 8; i += 512) {
    int c4 = i & 7;
    int pix = i >> 3;
    int xx = pix % PW1, r = pix / PW1;
    int xs = xx - 1, ys = y + r - 1;
    uint32_t v = 0;
    if (xs >= 0 && xs < 256 && ys >= 0 && ys < 256)
      v = *(const uint32_t*)(a1 + (((size_t)(b * 256 + ys) * 256 + xs) * 32) + c4 * 4);
    *(uint32_t*)(patch + pix * 32 + c4 * 4) = v;
  }
  __syncthreads();
  const int wave = tid >> 5, lane = tid & 31;
  const int p0 = wave * 16;
  const int m = lane & 15;
  const int px = p0 + m;
  const int off = (lane >> 4) * 8;

  v8i acc[4];
#pragma unroll
  for (int nt = 0; nt < 4; ++nt) acc[nt] = zero_v8i();
  const v8i* bp = (const v8i*)wpack;

  for (int kc = 0; kc < 5; ++kc) {
    int t0 = kc * 2, t1 = t0 + 1;
    int base0 = ((t0 / 3) * PW1 + px + (t0 % 3)) * 32;
    uint64_t q0 = *(const uint64_t*)(patch + base0 + off);
    uint64_t q1 = *(const uint64_t*)(patch + base0 + off + 16);
    uint64_t q2 = 0, q3 = 0;
    if (t1 < 9) {
      int base1 = ((t1 / 3) * PW1 + px + (t1 % 3)) * 32;
      q2 = *(const uint64_t*)(patch + base1 + off);
      q3 = *(const uint64_t*)(patch + base1 + off + 16);
    }
    v8i a = make_a(q0, q1, q2, q3);
#pragma unroll
    for (int nt = 0; nt < 4; ++nt)
      acc[nt] = wmma_iu8(a, bp[(kc * 4 + nt) * 32 + lane], acc[nt]);
  }

  float d1 = *d1p, d2 = *d2p;
#pragma unroll
  for (int nt = 0; nt < 4; ++nt) {
    int n = nt * 16 + m;
    float bt = ternf(b1[n], d1);
    float inv = gamma1[n] * rsqrtf(var1[n] + BN_EPS);
    float add = beta1[n] - mean1[n] * inv;
#pragma unroll
    for (int r = 0; r < 8; ++r) {
      int M = r + ((lane >> 4) << 3);
      float v = (float)acc[nt][r] + bt;
      v = v * inv + add;
      v = (v > 0.f) ? v : LRELU_SLOPE * v;
      int ox = p0 + M;
      a2[(((size_t)(b * 256 + y) * 256 + ox) * 64) + n] = (int8_t)ternf(v, d2);
    }
  }
}

#define PWC 148
__global__ __launch_bounds__(96) void conv2_fused_kernel(
    const int8_t* __restrict__ a2, const int8_t* __restrict__ xt,
    const int8_t* __restrict__ wpack2, const int8_t* __restrict__ wpack11,
    const float* __restrict__ d2p, const float* __restrict__ d11p,
    const float* __restrict__ b2, const float* __restrict__ b11,
    float* __restrict__ out) {
  __shared__ int8_t patch2[3 * PWC * 64];
  __shared__ int8_t patch1[3 * PWC * 32];
  const int half = blockIdx.x & 1;
  const int oy = (blockIdx.x >> 1) % 86;
  const int b = (blockIdx.x >> 1) / 86;
  const int tid = threadIdx.x;
  const int x0 = half * 48;
  const int ybase = 3 * oy - 1;
  const int xbase = 3 * x0 - 1;
  for (int i = tid; i < 3 * PWC * 16; i += 96) {
    int c4 = i & 15; int pix = i >> 4;
    int xx = pix % PWC, r = pix / PWC;
    int xs = xbase + xx, ys = ybase + r;
    uint32_t v = 0;
    if (xs >= 0 && xs < 256 && ys >= 0 && ys < 256)
      v = *(const uint32_t*)(a2 + (((size_t)(b * 256 + ys) * 256 + xs) * 64) + c4 * 4);
    *(uint32_t*)(patch2 + pix * 64 + c4 * 4) = v;
  }
  for (int i = tid; i < 3 * PWC * 8; i += 96) {
    int c4 = i & 7; int pix = i >> 3;
    int xx = pix % PWC, r = pix / PWC;
    int xs = xbase + xx, ys = ybase + r;
    uint32_t v = 0;
    if (xs >= 0 && xs < 256 && ys >= 0 && ys < 256)
      v = *(const uint32_t*)(xt + (((size_t)(b * 256 + ys) * 256 + xs) * 32) + c4 * 4);
    *(uint32_t*)(patch1 + pix * 32 + c4 * 4) = v;
  }
  __syncthreads();
  const int wave = tid >> 5, lane = tid & 31;
  const int p0 = x0 + wave * 16;
  const int m = lane & 15;
  int ox = p0 + m;
  int oxe = (ox < 86) ? ox : 85;
  const int lp = oxe - x0;
  const int off = (lane >> 4) * 8;

  v8i acc[4];
#pragma unroll
  for (int nt = 0; nt < 4; ++nt) acc[nt] = zero_v8i();
  const v8i* bp2 = (const v8i*)wpack2;
  const v8i* bp1 = (const v8i*)wpack11;

  for (int t = 0; t < 9; ++t) {
    int base = ((t / 3) * PWC + (3 * lp + (t % 3))) * 64;
    uint64_t q0 = *(const uint64_t*)(patch2 + base + off);
    uint64_t q1 = *(const uint64_t*)(patch2 + base + off + 16);
    uint64_t q2 = *(const uint64_t*)(patch2 + base + off + 32);
    uint64_t q3 = *(const uint64_t*)(patch2 + base + off + 48);
    v8i a = make_a(q0, q1, q2, q3);
#pragma unroll
    for (int nt = 0; nt < 4; ++nt)
      acc[nt] = wmma_iu8(a, bp2[(t * 4 + nt) * 32 + lane], acc[nt]);
  }
  for (int kc = 0; kc < 5; ++kc) {
    int t0 = kc * 2, t1 = t0 + 1;
    int base0 = ((t0 / 3) * PWC + (3 * lp + (t0 % 3))) * 32;
    uint64_t q0 = *(const uint64_t*)(patch1 + base0 + off);
    uint64_t q1 = *(const uint64_t*)(patch1 + base0 + off + 16);
    uint64_t q2 = 0, q3 = 0;
    if (t1 < 9) {
      int base1 = ((t1 / 3) * PWC + (3 * lp + (t1 % 3))) * 32;
      q2 = *(const uint64_t*)(patch1 + base1 + off);
      q3 = *(const uint64_t*)(patch1 + base1 + off + 16);
    }
    v8i a = make_a(q0, q1, q2, q3);
#pragma unroll
    for (int nt = 0; nt < 4; ++nt)
      acc[nt] = wmma_iu8(a, bp1[(kc * 4 + nt) * 32 + lane], acc[nt]);
  }

  float d2 = *d2p, d11 = *d11p;
#pragma unroll
  for (int nt = 0; nt < 4; ++nt) {
    int n = nt * 16 + m;
    float bt = ternf(b2[n], d2) + ternf(b11[n], d11);
#pragma unroll
    for (int r = 0; r < 8; ++r) {
      int M = r + ((lane >> 4) << 3);
      int oxx = p0 + M;
      if (oxx < 86) {
        float v = (float)acc[nt][r] + bt;
        v = (v > 0.f) ? v : LRELU_SLOPE * v;
        out[(((size_t)(b * 64 + n) * 86) + oy) * 86 + oxx] = v;
      }
    }
  }
}

extern "C" void kernel_launch(void* const* d_in, const int* in_sizes, int n_in,
                              void* d_out, int out_size, void* d_ws, size_t ws_size,
                              hipStream_t stream) {
  (void)in_sizes; (void)n_in; (void)out_size; (void)ws_size;
  const float* x       = (const float*)d_in[0];
  const float* w11     = (const float*)d_in[1];
  const float* b11     = (const float*)d_in[2];
  const float* w1      = (const float*)d_in[3];
  const float* b1      = (const float*)d_in[4];
  const float* w2      = (const float*)d_in[5];
  const float* b2      = (const float*)d_in[6];
  const float* gamma_p = (const float*)d_in[7];
  const float* beta_p  = (const float*)d_in[8];
  const float* mean_p  = (const float*)d_in[9];
  const float* var_p   = (const float*)d_in[10];
  const float* gamma1  = (const float*)d_in[11];
  const float* beta1   = (const float*)d_in[12];
  const float* mean1   = (const float*)d_in[13];
  const float* var1    = (const float*)d_in[14];
  float* out = (float*)d_out;

  char* ws = (char*)d_ws;
  float* d11p = (float*)(ws + 0);
  float* d1p  = (float*)(ws + 4);
  float* d2p  = (float*)(ws + 8);
  int8_t* wp11 = (int8_t*)(ws + 256);
  int8_t* wp1  = (int8_t*)(ws + 256 + 20480);
  int8_t* wp2  = (int8_t*)(ws + 256 + 40960);
  int8_t* xt   = (int8_t*)(ws + 78080);
  int8_t* a1   = xt + 33554432ull;
  int8_t* a2   = a1 + 33554432ull;

  maxabs_kernel<<<1, 256, 0, stream>>>(w11, 64 * 32 * 9, d11p);
  maxabs_kernel<<<1, 256, 0, stream>>>(w1,  64 * 32 * 9, d1p);
  maxabs_kernel<<<1, 256, 0, stream>>>(w2,  64 * 64 * 9, d2p);

  pack_w_kernel<<<40, 256, 0, stream>>>(w11, d11p, wp11, 32, 5);
  pack_w_kernel<<<40, 256, 0, stream>>>(w1,  d1p,  wp1,  32, 5);
  pack_w_kernel<<<72, 256, 0, stream>>>(w2,  d2p,  wp2,  64, 9);

  prep_act_kernel<<<16 * 256 * 4, 256, 0, stream>>>(x, d11p, d1p, gamma_p, beta_p,
                                                    mean_p, var_p, xt, a1);

  conv1_kernel<<<16 * 256, 512, 0, stream>>>(a1, wp1, d1p, d2p, b1,
                                             gamma1, beta1, mean1, var1, a2);

  conv2_fused_kernel<<<16 * 86 * 2, 96, 0, stream>>>(a2, xt, wp2, wp11,
                                                     d2p, d11p, b2, b11, out);
}
